// MultiHeadAttention_24000277250213
// MI455X (gfx1250) — matrix operand order repaired and checked by reading, not compiled, not run on hardware
//
#include <hip/hip_runtime.h>

typedef _Float16 v8h  __attribute__((ext_vector_type(8)));
typedef _Float16 v16h __attribute__((ext_vector_type(16)));
typedef float    v8f  __attribute__((ext_vector_type(8)));

#define B_  2
#define S_  4096
#define D_  1024
#define H_  16
#define HS_ 64
#define M_  (B_ * S_)   // 8192

__device__ __forceinline__ v16h cat8(v8h lo, v8h hi) {
  v16h r;
#pragma unroll
  for (int i = 0; i < 8; ++i) { r[i] = lo[i]; r[i + 8] = hi[i]; }
  return r;
}

__device__ __forceinline__ v8f zero8() {
  v8f r;
#pragma unroll
  for (int i = 0; i < 8; ++i) r[i] = 0.0f;
  return r;
}

__device__ __forceinline__ v8f wmma_f16(v16h a, v16h b, v8f c) {
  return __builtin_amdgcn_wmma_f32_16x16x32_f16(false, a, false, b, (short)0, c,
                                                false, false);
}

// CDNA5 async global->LDS copy (VGLOBAL encoding, tracked by ASYNCcnt).
// vdst = LDS byte address (low 32 bits of a generic shared pointer),
// vaddr = 64-bit global address, no saddr.
__device__ __forceinline__ void async_b128(const _Float16* g, _Float16* l) {
  unsigned laddr = (unsigned)(uintptr_t)l;
  asm volatile("global_load_async_to_lds_b128 %0, %1, off"
               :: "v"(laddr), "v"(g) : "memory");
}
__device__ __forceinline__ void wait_async0() {
  asm volatile("s_wait_asynccnt 0x0" ::: "memory");
}

// ---------------------------------------------------------------------------
// Prologue 1: x (fp32) -> xh (f16), contiguous.
// ---------------------------------------------------------------------------
__global__ __launch_bounds__(256) void convert_x(const float* __restrict__ X,
                                                 _Float16* __restrict__ Xh) {
  size_t i = ((size_t)blockIdx.x * 256 + threadIdx.x) * 8;
  float4 a = *(const float4*)(X + i);
  float4 b = *(const float4*)(X + i + 4);
  v8h o;
  o[0] = (_Float16)a.x; o[1] = (_Float16)a.y;
  o[2] = (_Float16)a.z; o[3] = (_Float16)a.w;
  o[4] = (_Float16)b.x; o[5] = (_Float16)b.y;
  o[6] = (_Float16)b.z; o[7] = (_Float16)b.w;
  *(v8h*)(Xh + i) = o;
}

// ---------------------------------------------------------------------------
// Prologue 2: W[k][n] fp32 -> Wt[n][k] f16 for all four weight matrices.
// 64x64 tiles via LDS; blockIdx.z selects the matrix.
// ---------------------------------------------------------------------------
__global__ __launch_bounds__(256) void transpose_w(
    const float* __restrict__ Wq, const float* __restrict__ Wk,
    const float* __restrict__ Wv, const float* __restrict__ Wo,
    _Float16* __restrict__ Wts) {
  __shared__ _Float16 sT[64 * 72];
  const int z = blockIdx.z;
  const float* W = (z == 0) ? Wq : (z == 1) ? Wk : (z == 2) ? Wv : Wo;
  _Float16* Out = Wts + (size_t)z * D_ * D_;
  const int k0 = blockIdx.x * 64, n0 = blockIdx.y * 64;
  const int tid = threadIdx.x;
#pragma unroll
  for (int q4 = 0; q4 < 4; ++q4) {
    int flat = (tid + 256 * q4) * 4;  // 0..4095
    int kr = flat >> 6, nc = flat & 63;
    const float4 f = *(const float4*)(W + (size_t)(k0 + kr) * D_ + n0 + nc);
    sT[(nc + 0) * 72 + kr] = (_Float16)f.x;
    sT[(nc + 1) * 72 + kr] = (_Float16)f.y;
    sT[(nc + 2) * 72 + kr] = (_Float16)f.z;
    sT[(nc + 3) * 72 + kr] = (_Float16)f.w;
  }
  __syncthreads();
#pragma unroll
  for (int q8 = 0; q8 < 2; ++q8) {
    int flat = (tid + 256 * q8) * 8;  // 0..4095
    int rn = flat >> 6, ck = flat & 63;
    *(v8h*)(Out + (size_t)(n0 + rn) * D_ + k0 + ck) =
        *(const v8h*)&sT[rn * 72 + ck];
  }
}

// ---------------------------------------------------------------------------
// Kernel 1: fused QKV projection, all-f16, async double-buffered tiles.
// Block tile 128x64, 8 waves (4x2), wave 32x32 via 2x2 WMMA, K-step 32.
// Q,K stored [B,H,S,HS]; V stored transposed [B,H,HS,S] (so the attention
// V tile becomes a contiguous row copy).
// ---------------------------------------------------------------------------
__global__ __launch_bounds__(256) void qkv_gemm(
    const _Float16* __restrict__ Xh, const _Float16* __restrict__ Wt,
    _Float16* __restrict__ Qo, _Float16* __restrict__ Ko,
    _Float16* __restrict__ Vo) {
  __shared__ _Float16 sA[2][128 * 40];  // [m][k], padded
  __shared__ _Float16 sB[2][64 * 40];   // [n][k], padded

  const int z = blockIdx.z;
  const _Float16* Wz = Wt + (size_t)z * D_ * D_;
  _Float16* O = (z == 0) ? Qo : ((z == 1) ? Ko : Vo);

  const int m0 = blockIdx.x * 128;
  const int n0 = blockIdx.y * 64;
  const int tid = threadIdx.x;
  const int wid = tid >> 5, lane = tid & 31;
  const int wm = wid & 3, wn = wid >> 2;
  const int lrow = lane & 15, g = lane >> 4;

  // per-thread staging coordinates (A: 2 chunks, B: 1 chunk of 8 halves)
  const int ar0 = (tid * 8) >> 5, ac0 = (tid * 8) & 31;
  const int ar1 = ((tid + 256) * 8) >> 5, ac1 = ((tid + 256) * 8) & 31;
  const int brn = (tid * 8) >> 5, bck = (tid * 8) & 31;

  v8f acc[2][2];
#pragma unroll
  for (int i = 0; i < 2; ++i)
#pragma unroll
    for (int j = 0; j < 2; ++j) acc[i][j] = zero8();

  // prime tile 0
  {
    async_b128(Xh + (size_t)(m0 + ar0) * D_ + ac0, &sA[0][ar0 * 40 + ac0]);
    async_b128(Xh + (size_t)(m0 + ar1) * D_ + ac1, &sA[0][ar1 * 40 + ac1]);
    async_b128(Wz + (size_t)(n0 + brn) * D_ + bck, &sB[0][brn * 40 + bck]);
  }
  wait_async0();
  __syncthreads();

  for (int it = 0; it < D_ / 32; ++it) {
    const int cur = it & 1;
    const int k0 = it * 32;
    if (it + 1 < D_ / 32) {
      const int kn = k0 + 32;
      async_b128(Xh + (size_t)(m0 + ar0) * D_ + kn + ac0,
                 &sA[cur ^ 1][ar0 * 40 + ac0]);
      async_b128(Xh + (size_t)(m0 + ar1) * D_ + kn + ac1,
                 &sA[cur ^ 1][ar1 * 40 + ac1]);
      async_b128(Wz + (size_t)(n0 + brn) * D_ + kn + bck,
                 &sB[cur ^ 1][brn * 40 + bck]);
    }

    v16h a[2], b[2];
#pragma unroll
    for (int mt = 0; mt < 2; ++mt) {
      int base = (wm * 32 + mt * 16 + lrow) * 40;
      v8h lo = *(const v8h*)&sA[cur][base + 8 * g];
      v8h hi = *(const v8h*)&sA[cur][base + 16 + 8 * g];
      a[mt] = cat8(lo, hi);
    }
#pragma unroll
    for (int nt = 0; nt < 2; ++nt) {
      int base = (wn * 32 + nt * 16 + lrow) * 40;
      v8h lo = *(const v8h*)&sB[cur][base + 8 * g];
      v8h hi = *(const v8h*)&sB[cur][base + 16 + 8 * g];
      b[nt] = cat8(lo, hi);
    }
#pragma unroll
    for (int mt = 0; mt < 2; ++mt)
#pragma unroll
      for (int nt = 0; nt < 2; ++nt)
        acc[mt][nt] = wmma_f16(a[mt], b[nt], acc[mt][nt]);

    wait_async0();
    __syncthreads();
  }

  // store f16; Q/K head-major [B,H,S,HS], V transposed [B,H,HS,S]
  const size_t ss = (z == 2) ? 1 : HS_;       // stride along s
  const size_t sh = (z == 2) ? (size_t)S_ : 1;  // stride along hs
#pragma unroll
  for (int mt = 0; mt < 2; ++mt)
#pragma unroll
    for (int nt = 0; nt < 2; ++nt)
#pragma unroll
      for (int r = 0; r < 8; ++r) {
        int m = m0 + wm * 32 + mt * 16 + r + 8 * g;
        int n = n0 + wn * 32 + nt * 16 + lrow;
        int bb = m >> 12, s = m & 4095;
        int h = n >> 6, hs = n & 63;
        O[(size_t)(bb * H_ + h) * (S_ * HS_) + (size_t)s * ss +
          (size_t)hs * sh] = (_Float16)acc[mt][nt][r];
      }
}

// ---------------------------------------------------------------------------
// Kernel 2: flash attention. One block per (b, h, 64-row q tile); 4 waves of
// 16 q-rows; K tile [kvrow][hs] and V tile [hs][kvrow] both staged by async
// copy with double buffering; online softmax in registers.
// ---------------------------------------------------------------------------
__global__ __launch_bounds__(128) void attn_kernel(
    const _Float16* __restrict__ Q, const _Float16* __restrict__ Kt,
    const _Float16* __restrict__ Vt, _Float16* __restrict__ Oout) {
  __shared__ _Float16 sK[2][64 * 72];  // [kvrow][hs] == B-frag layout (QK^T)
  __shared__ _Float16 sV[2][64 * 72];  // [hs][kvrow] == B-frag layout (P@V)
  __shared__ _Float16 sP[64 * 72];     // per-wave P staging, C->A relayout

  const int qt = blockIdx.x, h = blockIdx.y, b = blockIdx.z;
  const int tid = threadIdx.x;
  const int wid = tid >> 5, lane = tid & 31;
  const int lrow = lane & 15, g = lane >> 4;

  const size_t plane = (size_t)(b * H_ + h) * S_ * HS_;  // K plane
  const size_t vplane = plane;                            // V^T plane
  const int q0 = qt * 64 + wid * 16;

  // preload Q fragments (A layout, two 32-wide k chunks over HS)
  v16h qf[2];
#pragma unroll
  for (int kc = 0; kc < 2; ++kc) {
    const _Float16* p = Q + plane + (size_t)(q0 + lrow) * HS_ + kc * 32;
    v8h lo = *(const v8h*)(p + 8 * g);
    v8h hi = *(const v8h*)(p + 16 + 8 * g);
    qf[kc] = cat8(lo, hi);
  }

  float mrow[8], lsum[8];
  v8f oacc[4];
#pragma unroll
  for (int r = 0; r < 8; ++r) { mrow[r] = -1e30f; lsum[r] = 0.0f; }
#pragma unroll
  for (int j = 0; j < 4; ++j) oacc[j] = zero8();

  const float Cs = 0.125f * 1.44269504088896340736f;  // scale * log2(e)

  // per-thread staging coordinates: 4 K chunks + 4 V chunks of 8 halves
  int cr[4], cc[4];
#pragma unroll
  for (int q8 = 0; q8 < 4; ++q8) {
    int flat = (tid + 128 * q8) * 8;  // 0..4095
    cr[q8] = flat >> 6;
    cc[q8] = flat & 63;
  }

  // prime kv tile 0
#pragma unroll
  for (int q8 = 0; q8 < 4; ++q8) {
    async_b128(Kt + plane + (size_t)cr[q8] * HS_ + cc[q8],
               &sK[0][cr[q8] * 72 + cc[q8]]);
    async_b128(Vt + vplane + (size_t)cr[q8] * S_ + cc[q8],
               &sV[0][cr[q8] * 72 + cc[q8]]);
  }
  wait_async0();
  __syncthreads();

  for (int kv = 0; kv < S_ / 64; ++kv) {
    const int cur = kv & 1;
    if (kv + 1 < S_ / 64) {
      const int base = (kv + 1) * 64;
#pragma unroll
      for (int q8 = 0; q8 < 4; ++q8) {
        async_b128(Kt + plane + (size_t)(base + cr[q8]) * HS_ + cc[q8],
                   &sK[cur ^ 1][cr[q8] * 72 + cc[q8]]);
        async_b128(Vt + vplane + (size_t)cr[q8] * S_ + base + cc[q8],
                   &sV[cur ^ 1][cr[q8] * 72 + cc[q8]]);
      }
    }

    // scores: S = Q @ K^T (16x64 per wave)
    v8f sacc[4];
#pragma unroll
    for (int nt = 0; nt < 4; ++nt) {
      sacc[nt] = zero8();
#pragma unroll
      for (int kc = 0; kc < 2; ++kc) {
        int base = (nt * 16 + lrow) * 72 + kc * 32;
        v8h lo = *(const v8h*)&sK[cur][base + 8 * g];
        v8h hi = *(const v8h*)&sK[cur][base + 16 + 8 * g];
        sacc[nt] = wmma_f16(qf[kc], cat8(lo, hi), sacc[nt]);
      }
    }

    // online softmax (row = r + 8g)
    float mnew[8], alpha[8], rs[8];
#pragma unroll
    for (int r = 0; r < 8; ++r) {
      float v = sacc[0][r];
#pragma unroll
      for (int nt = 1; nt < 4; ++nt) v = fmaxf(v, sacc[nt][r]);
#pragma unroll
      for (int msk = 1; msk < 16; msk <<= 1)
        v = fmaxf(v, __shfl_xor(v, msk, 32));
      mnew[r] = fmaxf(mrow[r], v);
      alpha[r] = __builtin_amdgcn_exp2f((mrow[r] - mnew[r]) * Cs);
      rs[r] = 0.0f;
    }
#pragma unroll
    for (int nt = 0; nt < 4; ++nt)
#pragma unroll
      for (int r = 0; r < 8; ++r) {
        float p = __builtin_amdgcn_exp2f((sacc[nt][r] - mnew[r]) * Cs);
        rs[r] += p;
        sP[(wid * 16 + r + 8 * g) * 72 + lrow + 16 * nt] = (_Float16)p;
      }
#pragma unroll
    for (int r = 0; r < 8; ++r) {
#pragma unroll
      for (int msk = 1; msk < 16; msk <<= 1)
        rs[r] += __shfl_xor(rs[r], msk, 32);
      lsum[r] = lsum[r] * alpha[r] + rs[r];
      mrow[r] = mnew[r];
#pragma unroll
      for (int j = 0; j < 4; ++j) oacc[j][r] *= alpha[r];
    }
    __syncthreads();  // publish sP

    // O += P @ V
#pragma unroll
    for (int kc = 0; kc < 2; ++kc) {
      int abase = (wid * 16 + lrow) * 72 + kc * 32;
      v8h alo = *(const v8h*)&sP[abase + 8 * g];
      v8h ahi = *(const v8h*)&sP[abase + 16 + 8 * g];
      v16h pf = cat8(alo, ahi);
#pragma unroll
      for (int j = 0; j < 4; ++j) {
        int bbase = (j * 16 + lrow) * 72 + kc * 32;
        v8h blo = *(const v8h*)&sV[cur][bbase + 8 * g];
        v8h bhi = *(const v8h*)&sV[cur][bbase + 16 + 8 * g];
        oacc[j] = wmma_f16(pf, cat8(blo, bhi), oacc[j]);
      }
    }

    wait_async0();
    __syncthreads();  // next tile resident; cur buffer free for reuse
  }

  // normalize and store attn output f16 into [B,S,D], D index = h*64 + col
#pragma unroll
  for (int j = 0; j < 4; ++j)
#pragma unroll
    for (int r = 0; r < 8; ++r) {
      int row = q0 + r + 8 * g;
      int col = h * 64 + lrow + 16 * j;
      Oout[((size_t)(b * S_) + row) * D_ + col] =
          (_Float16)(oacc[j][r] / lsum[r]);
    }
}

// ---------------------------------------------------------------------------
// Kernel 3: output projection, all-f16 inputs, async double-buffered, fp32 out.
// ---------------------------------------------------------------------------
__global__ __launch_bounds__(256) void out_gemm(
    const _Float16* __restrict__ A, const _Float16* __restrict__ Wz,
    float* __restrict__ Out) {
  __shared__ _Float16 sA[2][128 * 40];
  __shared__ _Float16 sB[2][64 * 40];

  const int m0 = blockIdx.x * 128;
  const int n0 = blockIdx.y * 64;
  const int tid = threadIdx.x;
  const int wid = tid >> 5, lane = tid & 31;
  const int wm = wid & 3, wn = wid >> 2;
  const int lrow = lane & 15, g = lane >> 4;

  const int ar0 = (tid * 8) >> 5, ac0 = (tid * 8) & 31;
  const int ar1 = ((tid + 256) * 8) >> 5, ac1 = ((tid + 256) * 8) & 31;
  const int brn = (tid * 8) >> 5, bck = (tid * 8) & 31;

  v8f acc[2][2];
#pragma unroll
  for (int i = 0; i < 2; ++i)
#pragma unroll
    for (int j = 0; j < 2; ++j) acc[i][j] = zero8();

  async_b128(A + (size_t)(m0 + ar0) * D_ + ac0, &sA[0][ar0 * 40 + ac0]);
  async_b128(A + (size_t)(m0 + ar1) * D_ + ac1, &sA[0][ar1 * 40 + ac1]);
  async_b128(Wz + (size_t)(n0 + brn) * D_ + bck, &sB[0][brn * 40 + bck]);
  wait_async0();
  __syncthreads();

  for (int it = 0; it < D_ / 32; ++it) {
    const int cur = it & 1;
    if (it + 1 < D_ / 32) {
      const int kn = it * 32 + 32;
      async_b128(A + (size_t)(m0 + ar0) * D_ + kn + ac0,
                 &sA[cur ^ 1][ar0 * 40 + ac0]);
      async_b128(A + (size_t)(m0 + ar1) * D_ + kn + ac1,
                 &sA[cur ^ 1][ar1 * 40 + ac1]);
      async_b128(Wz + (size_t)(n0 + brn) * D_ + kn + bck,
                 &sB[cur ^ 1][brn * 40 + bck]);
    }

    v16h a[2], b[2];
#pragma unroll
    for (int mt = 0; mt < 2; ++mt) {
      int base = (wm * 32 + mt * 16 + lrow) * 40;
      v8h lo = *(const v8h*)&sA[cur][base + 8 * g];
      v8h hi = *(const v8h*)&sA[cur][base + 16 + 8 * g];
      a[mt] = cat8(lo, hi);
    }
#pragma unroll
    for (int nt = 0; nt < 2; ++nt) {
      int base = (wn * 32 + nt * 16 + lrow) * 40;
      v8h lo = *(const v8h*)&sB[cur][base + 8 * g];
      v8h hi = *(const v8h*)&sB[cur][base + 16 + 8 * g];
      b[nt] = cat8(lo, hi);
    }
#pragma unroll
    for (int mt = 0; mt < 2; ++mt)
#pragma unroll
      for (int nt = 0; nt < 2; ++nt)
        acc[mt][nt] = wmma_f16(a[mt], b[nt], acc[mt][nt]);

    wait_async0();
    __syncthreads();
  }

#pragma unroll
  for (int mt = 0; mt < 2; ++mt)
#pragma unroll
    for (int nt = 0; nt < 2; ++nt)
#pragma unroll
      for (int r = 0; r < 8; ++r) {
        int m = m0 + wm * 32 + mt * 16 + r + 8 * g;
        int n = n0 + wn * 32 + nt * 16 + lrow;
        Out[(size_t)m * D_ + n] = acc[mt][nt][r];
      }
}

// ---------------------------------------------------------------------------
extern "C" void kernel_launch(void* const* d_in, const int* in_sizes, int n_in,
                              void* d_out, int out_size, void* d_ws,
                              size_t ws_size, hipStream_t stream) {
  const float* x  = (const float*)d_in[0];
  const float* Wq = (const float*)d_in[1];
  const float* Wk = (const float*)d_in[2];
  const float* Wv = (const float*)d_in[3];
  const float* Wo = (const float*)d_in[4];
  float* out = (float*)d_out;

  // workspace layout (f16 elements):
  //   xh: 8M | Wt: 4x1M | Q: 8M | K: 8M | Vt: 8M | AO: 8M  => 88 MB total
  _Float16* xh = (_Float16*)d_ws;
  _Float16* wt = xh + (size_t)M_ * D_;
  _Float16* q  = wt + (size_t)4 * D_ * D_;
  _Float16* k  = q + (size_t)M_ * D_;
  _Float16* vt = k + (size_t)M_ * D_;
  _Float16* ao = vt + (size_t)M_ * D_;

  convert_x<<<dim3((M_ * D_) / (256 * 8)), 256, 0, stream>>>(x, xh);
  transpose_w<<<dim3(D_ / 64, D_ / 64, 4), 256, 0, stream>>>(Wq, Wk, Wv, Wo,
                                                             wt);

  qkv_gemm<<<dim3(M_ / 128, D_ / 64, 3), 256, 0, stream>>>(xh, wt, q, k, vt);

  attn_kernel<<<dim3(S_ / 64, H_, B_), 128, 0, stream>>>(q, k, vt, ao);

  out_gemm<<<dim3(M_ / 128, D_ / 64), 256, 0, stream>>>(
      ao, wt + (size_t)3 * D_ * D_, out);
}
